// MambaLateralConnection_10857677324810
// MI455X (gfx1250) — matrix operand order repaired and checked by reading, not compiled, not run on hardware
//
#include <hip/hip_runtime.h>
#include <hip/hip_bf16.h>

// ---------------------------------------------------------------------------
// MambaLateralConnection on gfx1250 (MI455X).
// - All GEMM-shaped ops -> v_wmma_f32_16x16x32_bf16 (bf16 in, fp32 accum).
// - Fast-path GEMM: double-buffered LDS; B tiles staged by the Tensor Data
//   Mover (tensor_load_to_lds / s_wait_tensorcnt), A tiles by
//   global_load_async_to_lds_b128 (s_wait_asynccnt); DMA for step k+1
//   overlaps WMMA of step k.
// - 128x64x32 block tile, 8 waves, 4 WMMAs per wave per K-step.
// ---------------------------------------------------------------------------

typedef __bf16 bh16;
typedef unsigned int u32;
typedef __attribute__((ext_vector_type(16))) __bf16 v16bf;
typedef __attribute__((ext_vector_type(8)))  float  v8f;
typedef __attribute__((ext_vector_type(4)))  unsigned int v4u;
typedef __attribute__((ext_vector_type(8)))  unsigned int v8u;

#define BM 128
#define BN 64
#define BK 32
#define ASTR 48   // LDS row stride (bf16) for A tile: keeps b128 alignment
#define BSTR 48   // LDS row stride for transposed B tile

// Problem constants
#define NB   8
#define HW   4096
#define CIN  512
#define COUT 256
#define DIN  512
#define E2   1024
#define NPRJ 48

__device__ __forceinline__ v8f wmma_bf16(v16bf a, v16bf b, v8f c) {
  return __builtin_amdgcn_wmma_f32_16x16x32_bf16(false, a, false, b, (short)0, c,
                                                 false, false);
}

__device__ __forceinline__ v8f zero_v8f() {
  v8f z;
#pragma unroll
  for (int i = 0; i < 8; ++i) z[i] = 0.0f;
  return z;
}

__device__ __forceinline__ void wait_async0() {
#if __has_builtin(__builtin_amdgcn_s_wait_asynccnt)
  __builtin_amdgcn_s_wait_asynccnt(0);
#else
  asm volatile("s_wait_asynccnt 0x0" ::: "memory");
#endif
}

// Per-lane async copy of 16B global -> LDS (tracked by ASYNCcnt).
__device__ __forceinline__ void async_g2l_b128(u32 lds_byte_addr, const void* g) {
  asm volatile("global_load_async_to_lds_b128 %0, %1, off"
               :: "v"(lds_byte_addr), "v"(g)
               : "memory");
}

// TDM 2-D tile load: tile_x elems (2B each) x tile_y rows, row stride
// stride_elems, landing contiguously at lds_byte_addr. Issued once per wave.
__device__ __forceinline__ void tdm_load_2d(const void* gaddr, u32 lds_byte_addr,
                                            u32 tile_x, u32 tile_y,
                                            u32 stride_elems) {
  unsigned long long ga = (unsigned long long)(size_t)gaddr;
  v4u g0;
  g0[0] = 1u;                                           // count=1, user mode
  g0[1] = lds_byte_addr;                                // lds_addr
  g0[2] = (u32)(ga & 0xffffffffu);                      // global_addr[31:0]
  g0[3] = (u32)((ga >> 32) & 0x01ffffffu) | (2u << 30); // addr[56:32], type=2
  v8u g1;
  g1[0] = (1u << 16);                                   // data_size=1 (2 bytes)
  g1[1] = (tile_x & 0xffffu) << 16;                     // tensor_dim0[15:0]
  g1[2] = ((tile_x >> 16) & 0xffffu) | ((tile_y & 0xffffu) << 16);
  g1[3] = ((tile_y >> 16) & 0xffffu) | ((tile_x & 0xffffu) << 16); // tile_dim0
  g1[4] = (tile_y & 0xffffu);                           // tile_dim1
  g1[5] = stride_elems;                                 // dim0 stride lo32
  g1[6] = 0u;
  g1[7] = 0u;
  asm volatile("tensor_load_to_lds %0, %1" :: "s"(g0), "s"(g1) : "memory");
}

// Fragment gathers per documented 16-bit WMMA VGPR layouts.
__device__ __forceinline__ v16bf lds_frag_a(const bh16* As, int mt, int lane) {
  const int row = lane & 15, kh = (lane >> 4) * 8;
  v16bf f;
  const bh16* p = &As[(mt * 16 + row) * ASTR + kh];
  ((uint4*)&f)[0] = *(const uint4*)(p);
  ((uint4*)&f)[1] = *(const uint4*)(p + 16);
  return f;
}
__device__ __forceinline__ v16bf lds_frag_b(const bh16* Bs, int nt, int lane) {
  const int col = lane & 15, kb = (lane >> 4) * 8;
  v16bf f;
  const bh16* p = &Bs[(nt * 16 + col) * BSTR + kb];
  ((uint4*)&f)[0] = *(const uint4*)(p);
  ((uint4*)&f)[1] = *(const uint4*)(p + 16);
  return f;
}

// ---------------------------------------------------------------------------
// Batched WMMA GEMM: C[b] = A(MxK bf16 row-major) * B[b](KxN bf16 row-major).
// GUARDED=false requires K%32==0 (TDM + async staging, double buffered).
// ---------------------------------------------------------------------------
template <bool GUARDED>
__global__ __launch_bounds__(256)
void gemm_bf16_wmma(const bh16* __restrict__ A, const bh16* __restrict__ B,
                    float* __restrict__ Cf, bh16* __restrict__ Cb,
                    const float* __restrict__ bias,
                    const float* __restrict__ addsrc,
                    int M, int N, int K, int lda, int ldb, int ldc,
                    long long strideB, long long strideC)
{
  __shared__ bh16 As[2][BM * ASTR];
  __shared__ bh16 Braw[2][BK * BN];   // TDM landing tiles (row-major)
  __shared__ bh16 Bs[2][BN * BSTR];   // transposed for fragment loads

  const int tid  = threadIdx.x;
  const int lane = tid & 31;
  const int wave = tid >> 5;
  const int bz   = blockIdx.z;

  B += (long long)bz * strideB;
  const long long coff = (long long)bz * strideC;

  const int m0 = blockIdx.y * BM;
  const int n0 = blockIdx.x * BN;

  const int mg = wave & 3;   // m-tile pair: tiles 2*mg, 2*mg+1
  const int ng = wave >> 2;  // n-tile pair: tiles 2*ng, 2*ng+1

  v8f acc00 = zero_v8f(), acc01 = zero_v8f();
  v8f acc10 = zero_v8f(), acc11 = zero_v8f();

  // A staging: 128x32 tile = 512 16B-chunks; each thread owns 2 chunks.
  const int r0 = tid >> 2;          // 0..63 (second chunk: +64)
  const int kq = (tid & 3) * 8;     // 0,8,16,24
  // B staging: 32x64 tile; thread -> (k-row, 8-col chunk)
  const int bk = tid >> 3;          // 0..31
  const int bn = (tid & 7) * 8;     // 0..56

  u32 ldsA0b[2], ldsA1b[2], ldsBrawb[2];
#pragma unroll
  for (int s = 0; s < 2; ++s) {
    ldsA0b[s]   = (u32)(size_t)(&As[s][r0 * ASTR + kq]);
    ldsA1b[s]   = (u32)(size_t)(&As[s][(r0 + 64) * ASTR + kq]);
    ldsBrawb[s] = (u32)(size_t)(&Braw[s][0]);
  }
  const bh16* gA0 = A + (long long)(m0 + r0) * lda + kq;
  const bh16* gA1 = A + (long long)(m0 + r0 + 64) * lda + kq;

  if constexpr (!GUARDED) {
    auto stage = [&](int k0s, int s) {
      if (wave == 0)
        tdm_load_2d(B + (long long)k0s * ldb + n0, ldsBrawb[s], BN, BK, (u32)ldb);
      async_g2l_b128(ldsA0b[s], gA0 + k0s);
      async_g2l_b128(ldsA1b[s], gA1 + k0s);
    };
    stage(0, 0);
    int buf = 0;
    for (int k0 = 0; k0 < K; k0 += BK, buf ^= 1) {
      // retire stage k0 (only these are outstanding for this wave)
      wait_async0();
      if (wave == 0) __builtin_amdgcn_s_wait_tensorcnt(0);
      __syncthreads();  // B1: stage k0 in LDS; step k0-BK reads all retired
      // overlap: start DMA for the next K-step into the other buffer
      if (k0 + BK < K) stage(k0 + BK, buf ^ 1);
      // transpose Braw[buf][k][n] -> Bs[buf][n][k]
      uint4 bv = *(const uint4*)(&Braw[buf][bk * BN + bn]);
      union { uint4 u; bh16 h[8]; } bu; bu.u = bv;
#pragma unroll
      for (int j = 0; j < 8; ++j) Bs[buf][(bn + j) * BSTR + bk] = bu.h[j];
      __syncthreads();  // B2: Bs[buf] ready

      v16bf a0 = lds_frag_a(&As[buf][0], mg * 2 + 0, lane);
      v16bf a1 = lds_frag_a(&As[buf][0], mg * 2 + 1, lane);
      v16bf b0 = lds_frag_b(&Bs[buf][0], ng * 2 + 0, lane);
      v16bf b1 = lds_frag_b(&Bs[buf][0], ng * 2 + 1, lane);
      acc00 = wmma_bf16(a0, b0, acc00);
      acc01 = wmma_bf16(a0, b1, acc01);
      acc10 = wmma_bf16(a1, b0, acc10);
      acc11 = wmma_bf16(a1, b1, acc11);
    }
  } else {
    for (int k0 = 0; k0 < K; k0 += BK) {
      // branchless clamp + select staging (single buffer)
#pragma unroll
      for (int c = 0; c < 2; ++c) {
        const int mm = m0 + r0 + c * 64, kk = k0 + kq;
        const int mcl = mm < M ? mm : (M - 1);
        const int kcl = (kk + 8) <= K ? kk : 0;
        uint4 av = *(const uint4*)(A + (long long)mcl * lda + kcl);
        if (mm >= M || kk >= K) av = make_uint4(0u, 0u, 0u, 0u);
        *(uint4*)(&As[0][(r0 + c * 64) * ASTR + kq]) = av;
      }
      const int kr  = k0 + bk;
      const int krc = kr < K ? kr : 0;
      uint4 bv = *(const uint4*)(B + (long long)krc * ldb + (n0 + bn));
      if (kr >= K) bv = make_uint4(0u, 0u, 0u, 0u);
      union { uint4 u; bh16 h[8]; } bu; bu.u = bv;
#pragma unroll
      for (int j = 0; j < 8; ++j) Bs[0][(bn + j) * BSTR + bk] = bu.h[j];
      __syncthreads();

      v16bf a0 = lds_frag_a(&As[0][0], mg * 2 + 0, lane);
      v16bf a1 = lds_frag_a(&As[0][0], mg * 2 + 1, lane);
      v16bf b0 = lds_frag_b(&Bs[0][0], ng * 2 + 0, lane);
      v16bf b1 = lds_frag_b(&Bs[0][0], ng * 2 + 1, lane);
      acc00 = wmma_bf16(a0, b0, acc00);
      acc01 = wmma_bf16(a0, b1, acc01);
      acc10 = wmma_bf16(a1, b0, acc10);
      acc11 = wmma_bf16(a1, b1, acc11);
      __syncthreads();
    }
  }

  const int col = lane & 15;
  const int rb  = (lane >> 4) * 8;
  const int gcol0 = n0 + (ng * 2 + 0) * 16 + col;
  const int gcol1 = n0 + (ng * 2 + 1) * 16 + col;

  auto emit = [&](float val, int grow, int gcol) {
    if (grow >= M) return;
    float x = val;
    if (bias) x += bias[grow];
    long long idx = coff + (long long)grow * ldc + gcol;
    if (addsrc) x += addsrc[idx];
    if (Cf) Cf[idx] = x;
    if (Cb) Cb[idx] = (bh16)x;
  };
#pragma unroll
  for (int v = 0; v < 8; ++v) {
    int g0r = m0 + (mg * 2 + 0) * 16 + rb + v;
    int g1r = m0 + (mg * 2 + 1) * 16 + rb + v;
    emit(acc00[v], g0r, gcol0);
    emit(acc01[v], g0r, gcol1);
    emit(acc10[v], g1r, gcol0);
    emit(acc11[v], g1r, gcol1);
  }
}

// ---------------------------------------------------------------------------
// 3x3 refine conv: 9-tap accumulated implicit WMMA GEMM, fused BN + ReLU.
// A (weights) staged via async-to-LDS; B staged branchlessly (clamp+select).
// ---------------------------------------------------------------------------
__global__ __launch_bounds__(256)
void refine_wmma(const bh16* __restrict__ Wp,   // packed [tap][256][256]
                 const bh16* __restrict__ Xr,   // [b][256][4096]
                 const float* __restrict__ rbias,
                 const float* __restrict__ gma, const float* __restrict__ bta,
                 const float* __restrict__ mean, const float* __restrict__ var,
                 float* __restrict__ out)
{
  __shared__ bh16 As[BM * ASTR];
  __shared__ bh16 Bs[BN * BSTR];

  const int tid  = threadIdx.x;
  const int lane = tid & 31;
  const int wave = tid >> 5;

  const int h  = blockIdx.x;       // image row; tile N == full row (64 px)
  const int m0 = blockIdx.y * BM;
  const int bz = blockIdx.z;

  const bh16* X = Xr + (long long)bz * COUT * HW;

  const int mg = wave & 3;
  const int ng = wave >> 2;

  v8f acc00 = zero_v8f(), acc01 = zero_v8f();
  v8f acc10 = zero_v8f(), acc11 = zero_v8f();

  const int r0 = tid >> 2;
  const int kq = (tid & 3) * 8;
  const int bk = tid >> 3;
  const int bn = (tid & 7) * 8;

  const u32 ldsA0 = (u32)(size_t)(&As[r0 * ASTR + kq]);
  const u32 ldsA1 = (u32)(size_t)(&As[(r0 + 64) * ASTR + kq]);

  for (int tap = 0; tap < 9; ++tap) {
    const int ky = tap / 3, kx = tap % 3;
    const int hh = h + ky - 1;
    const bool hok = (hh >= 0) && (hh < 64);
    const int  hc  = hok ? hh : 0;
    const bh16* Wt = Wp + (long long)tap * COUT * COUT;

    for (int k0 = 0; k0 < COUT; k0 += BK) {
      __syncthreads();  // close previous iteration's LDS reads
      async_g2l_b128(ldsA0, Wt + (long long)(m0 + r0) * COUT + (k0 + kq));
      async_g2l_b128(ldsA1, Wt + (long long)(m0 + r0 + 64) * COUT + (k0 + kq));
      const bh16* Xrow = X + (long long)(k0 + bk) * HW + hc * 64;
#pragma unroll
      for (int j = 0; j < 8; ++j) {
        int w = bn + j + kx - 1;
        bool wok = (w >= 0) && (w < 64);
        int wc = wok ? w : 0;
        bh16 v = Xrow[wc];               // unconditional (clamped) load
        if (!(hok && wok)) v = (bh16)0.0f;
        Bs[(bn + j) * BSTR + bk] = v;
      }
      wait_async0();
      __syncthreads();

      v16bf a0 = lds_frag_a(As, mg * 2 + 0, lane);
      v16bf a1 = lds_frag_a(As, mg * 2 + 1, lane);
      v16bf b0 = lds_frag_b(Bs, ng * 2 + 0, lane);
      v16bf b1 = lds_frag_b(Bs, ng * 2 + 1, lane);
      acc00 = wmma_bf16(a0, b0, acc00);
      acc01 = wmma_bf16(a0, b1, acc01);
      acc10 = wmma_bf16(a1, b0, acc10);
      acc11 = wmma_bf16(a1, b1, acc11);
    }
  }

  const int col = lane & 15;
  const int rb  = (lane >> 4) * 8;
  const int gcol0 = (ng * 2 + 0) * 16 + col;
  const int gcol1 = (ng * 2 + 1) * 16 + col;

  auto emit = [&](float val, int grow, int gcol) {
    float inv = gma[grow] * rsqrtf(var[grow] + 1e-5f);
    float x = (val + rbias[grow] - mean[grow]) * inv + bta[grow];
    out[(long long)bz * COUT * HW + (long long)grow * HW + h * 64 + gcol] =
        fmaxf(x, 0.0f);
  };
#pragma unroll
  for (int v = 0; v < 8; ++v) {
    int g0r = m0 + (mg * 2 + 0) * 16 + rb + v;
    int g1r = m0 + (mg * 2 + 1) * 16 + rb + v;
    emit(acc00[v], g0r, gcol0);
    emit(acc01[v], g0r, gcol1);
    emit(acc10[v], g1r, gcol0);
    emit(acc11[v], g1r, gcol1);
  }
}

// ---------------------------------------------------------------------------
// Depthwise causal conv1d (k=4) + SiLU over xm (= rows 0..511 of xz).
// ---------------------------------------------------------------------------
__global__ __launch_bounds__(256)
void dwconv_silu(const bh16* __restrict__ xz, const float* __restrict__ w,
                 const float* __restrict__ bias,
                 float* __restrict__ xcf, bh16* __restrict__ xcb)
{
  const int t = blockIdx.x * 256 + threadIdx.x;
  const int d = blockIdx.y;
  const int b = blockIdx.z;
  const bh16* src = xz + ((long long)b * E2 + d) * HW;
  float acc = bias[d];
#pragma unroll
  for (int j = 0; j < 4; ++j) {
    int tt = t - 3 + j;
    if (tt >= 0) acc += w[d * 4 + j] * (float)src[tt];
  }
  float s = acc / (1.0f + __expf(-acc));
  long long idx = ((long long)b * DIN + d) * HW + t;
  xcf[idx] = s;
  xcb[idx] = (bh16)s;
}

// ---------------------------------------------------------------------------
// Selective scan: one thread per (batch, channel), 16 SSM states in VGPRs,
// B/C SSM vectors staged per 64-step chunk in LDS.
// ---------------------------------------------------------------------------
__global__ __launch_bounds__(512)
void scan_kernel(const float* __restrict__ dtl, const float* __restrict__ proj,
                 const float* __restrict__ xc, const bh16* __restrict__ xz,
                 const float* __restrict__ A_log, const float* __restrict__ Dp,
                 bh16* __restrict__ yb)
{
  const int d = threadIdx.x;
  const int b = blockIdx.x;
  __shared__ float Bs[16][64];
  __shared__ float Cs[16][64];

  float Arow[16];
#pragma unroll
  for (int n = 0; n < 16; ++n) Arow[n] = -__expf(A_log[d * 16 + n]);
  const float Dd = Dp[d];

  const long long base = ((long long)b * DIN + d) * HW;
  const float* dptr = dtl + base;
  const float* xptr = xc + base;
  const bh16*  zptr = xz + ((long long)b * E2 + DIN + d) * HW;
  const float* Bg = proj + ((long long)b * NPRJ + 16) * HW;
  const float* Cg = proj + ((long long)b * NPRJ + 32) * HW;

  float hst[16];
#pragma unroll
  for (int n = 0; n < 16; ++n) hst[n] = 0.0f;

  for (int t0 = 0; t0 < HW; t0 += 64) {
    __syncthreads();
    for (int i = d; i < 1024; i += 512) {
      int n = i >> 6, tc = i & 63;
      Bs[n][tc] = Bg[(long long)n * HW + t0 + tc];
      Cs[n][tc] = Cg[(long long)n * HW + t0 + tc];
    }
    __syncthreads();
    for (int tc = 0; tc < 64; ++tc) {
      const int t = t0 + tc;
      float xdt = dptr[t];
      float dt  = (xdt > 20.0f) ? xdt : __logf(1.0f + __expf(xdt));
      float xv  = xptr[t];
      float acc = 0.0f;
#pragma unroll
      for (int n = 0; n < 16; ++n) {
        float dA = __expf(dt * Arow[n]);
        hst[n] = dA * hst[n] + (dt * Bs[n][tc]) * xv;
        acc += hst[n] * Cs[n][tc];
      }
      float zv = (float)zptr[t];
      float sz = zv / (1.0f + __expf(-zv));
      yb[base + t] = (bh16)((acc + xv * Dd) * sz);
    }
  }
}

// ---------------------------------------------------------------------------
__global__ void cvt_f32_bf16(const float* __restrict__ s, bh16* __restrict__ d,
                             long long n)
{
  long long i = (long long)blockIdx.x * blockDim.x + threadIdx.x;
  if (i < n) d[i] = (bh16)s[i];
}

__global__ void pack_refine(const float* __restrict__ s, bh16* __restrict__ d)
{
  int i = blockIdx.x * 256 + threadIdx.x;
  if (i >= COUT * COUT * 9) return;
  int o = i / (COUT * 9);
  int r = i % (COUT * 9);
  int ci = r / 9;
  int t  = r % 9;
  d[(long long)t * COUT * COUT + o * COUT + ci] = (bh16)s[i];
}

// ---------------------------------------------------------------------------
extern "C" void kernel_launch(void* const* d_in, const int* in_sizes, int n_in,
                              void* d_out, int out_size, void* d_ws, size_t ws_size,
                              hipStream_t stream)
{
  (void)in_sizes; (void)n_in; (void)out_size; (void)ws_size;

  const float* x         = (const float*)d_in[0];
  const float* lateral_w = (const float*)d_in[1];
  const float* lateral_b = (const float*)d_in[2];
  const float* in_proj_w = (const float*)d_in[3];
  const float* conv1d_w  = (const float*)d_in[4];
  const float* conv1d_b  = (const float*)d_in[5];
  const float* x_proj_w  = (const float*)d_in[6];
  const float* dt_proj_w = (const float*)d_in[7];
  const float* dt_proj_b = (const float*)d_in[8];
  const float* A_log     = (const float*)d_in[9];
  const float* D_param   = (const float*)d_in[10];
  const float* out_proj_w= (const float*)d_in[11];
  const float* refine_w  = (const float*)d_in[12];
  const float* refine_b  = (const float*)d_in[13];
  const float* bn_gamma  = (const float*)d_in[14];
  const float* bn_beta   = (const float*)d_in[15];
  const float* bn_mean   = (const float*)d_in[16];
  const float* bn_var    = (const float*)d_in[17];
  float* out = (float*)d_out;

  size_t off = 0;
  char* ws = (char*)d_ws;
  auto carve = [&](size_t bytes) {
    void* p = ws + off;
    off += (bytes + 255) & ~(size_t)255;
    return p;
  };
  const long long nXB  = (long long)NB * CIN * HW;
  const long long nXL  = (long long)NB * COUT * HW;
  const long long nXZ  = (long long)NB * E2 * HW;
  const long long nXC  = (long long)NB * DIN * HW;
  const long long nPRJ = (long long)NB * NPRJ * HW;

  bh16*  xb     = (bh16*) carve(nXB * 2);
  bh16*  wl_b   = (bh16*) carve((size_t)COUT * CIN * 2);
  bh16*  wip_b  = (bh16*) carve((size_t)E2 * COUT * 2);
  bh16*  wxp_b  = (bh16*) carve((size_t)NPRJ * DIN * 2);
  bh16*  wdt_b  = (bh16*) carve((size_t)DIN * 16 * 2);
  bh16*  wout_b = (bh16*) carve((size_t)COUT * DIN * 2);
  bh16*  wref_b = (bh16*) carve((size_t)9 * COUT * COUT * 2);
  float* xl_f   = (float*) carve(nXL * 4);
  bh16*  xl_b   = (bh16*)  carve(nXL * 2);
  bh16*  xz_b   = (bh16*)  carve(nXZ * 2);
  float* xc_f   = (float*) carve(nXC * 4);
  bh16*  xc_b   = (bh16*)  carve(nXC * 2);
  float* prj_f  = (float*) carve(nPRJ * 4);
  bh16*  prj_b  = (bh16*)  carve(nPRJ * 2);
  float* dtl_f  = (float*) carve(nXC * 4);
  bh16*  y_b    = (bh16*)  carve(nXC * 2);
  bh16*  xr_b   = (bh16*)  carve(nXL * 2);

  auto cvt = [&](const float* s, bh16* d, long long n) {
    int blocks = (int)((n + 255) / 256);
    cvt_f32_bf16<<<blocks, 256, 0, stream>>>(s, d, n);
  };
  cvt(x, xb, nXB);
  cvt(lateral_w,  wl_b,   (long long)COUT * CIN);
  cvt(in_proj_w,  wip_b,  (long long)E2 * COUT);
  cvt(x_proj_w,   wxp_b,  (long long)NPRJ * DIN);
  cvt(dt_proj_w,  wdt_b,  (long long)DIN * 16);
  cvt(out_proj_w, wout_b, (long long)COUT * DIN);
  pack_refine<<<(COUT * COUT * 9 + 255) / 256, 256, 0, stream>>>(refine_w, wref_b);

  const dim3 blk(256);

  // 1) lateral: xl = W_l * x + b
  gemm_bf16_wmma<false><<<dim3(HW / BN, COUT / BM, NB), blk, 0, stream>>>(
      wl_b, xb, xl_f, xl_b, lateral_b, nullptr,
      COUT, HW, CIN, CIN, HW, HW,
      (long long)CIN * HW, (long long)COUT * HW);

  // 2) in_proj: xz = W_in * xl
  gemm_bf16_wmma<false><<<dim3(HW / BN, E2 / BM, NB), blk, 0, stream>>>(
      wip_b, xl_b, nullptr, xz_b, nullptr, nullptr,
      E2, HW, COUT, COUT, HW, HW,
      (long long)COUT * HW, (long long)E2 * HW);

  // 3) depthwise causal conv + SiLU
  dwconv_silu<<<dim3(HW / 256, DIN, NB), blk, 0, stream>>>(
      xz_b, conv1d_w, conv1d_b, xc_f, xc_b);

  // 4) x_proj (M=48: guarded)
  gemm_bf16_wmma<true><<<dim3(HW / BN, (NPRJ + BM - 1) / BM, NB), blk, 0, stream>>>(
      wxp_b, xc_b, prj_f, prj_b, nullptr, nullptr,
      NPRJ, HW, DIN, DIN, HW, HW,
      (long long)DIN * HW, (long long)NPRJ * HW);

  // 5) dt (K=16: guarded): dtlin = W_dt * dt_r + dt_bias
  gemm_bf16_wmma<true><<<dim3(HW / BN, DIN / BM, NB), blk, 0, stream>>>(
      wdt_b, prj_b, dtl_f, nullptr, dt_proj_b, nullptr,
      DIN, HW, 16, 16, HW, HW,
      (long long)NPRJ * HW, (long long)DIN * HW);

  // 6) selective scan
  scan_kernel<<<dim3(NB), dim3(512), 0, stream>>>(
      dtl_f, prj_f, xc_f, xz_b, A_log, D_param, y_b);

  // 7) out_proj + residual: xr = xl + W_out * y
  gemm_bf16_wmma<false><<<dim3(HW / BN, COUT / BM, NB), blk, 0, stream>>>(
      wout_b, y_b, nullptr, xr_b, nullptr, xl_f,
      COUT, HW, DIN, DIN, HW, HW,
      (long long)DIN * HW, (long long)COUT * HW);

  // 8) 3x3 refine conv + BN + ReLU
  refine_wmma<<<dim3(64, COUT / BM, NB), blk, 0, stream>>>(
      wref_b, xr_b, refine_b, bn_gamma, bn_beta, bn_mean, bn_var, out);
}
